// MixedRNNBlock_47717086659296
// MI455X (gfx1250) — matrix operand order repaired and checked by reading, not compiled, not run on hardware
//
#include <hip/hip_runtime.h>
#include <stdint.h>

// ---------------- problem constants ----------------
#define Bb   32
#define Tt   512
#define Ii   512
#define Hh   512
#define GCOLS 7168            // 2*4H + 2*3H = 2*2048 + 2*1536
#define NWG_REC 32
#define G_LDS_OFF 262144u     // byte offset of the G-slice staging buffer in LDS

typedef __bf16        v16bf __attribute__((ext_vector_type(16)));
typedef float         v8f   __attribute__((ext_vector_type(8)));
typedef unsigned int  v4u   __attribute__((ext_vector_type(4)));
typedef int           v8i   __attribute__((ext_vector_type(8)));
typedef int           v4i   __attribute__((ext_vector_type(4)));

#define GAS __attribute__((address_space(1)))
typedef const GAS v4u* gv4cp;            // explicit global pointer -> global_load_b128

union Frag32 { v4u u[2]; v16bf v; };     // 32 bytes = 16 bf16 per lane

#ifndef __has_builtin
#define __has_builtin(x) 0
#endif
#if __has_builtin(__builtin_amdgcn_tensor_load_to_lds) && \
    __has_builtin(__builtin_amdgcn_s_wait_tensorcnt)
#define HAVE_TDM 1
#else
#define HAVE_TDM 0
#endif

__device__ __forceinline__ unsigned short f2bf(float f) {
  union { float f; unsigned u; } cv; cv.f = f;
  return (unsigned short)((cv.u + 0x7FFFu + ((cv.u >> 16) & 1u)) >> 16);
}
__device__ __forceinline__ float sigm(float x) { return 1.0f / (1.0f + __expf(-x)); }
__device__ __forceinline__ float tanhfast(float x) {
  float e = __expf(-2.0f * x);
  return (1.0f - e) / (1.0f + e);
}

#define WMMA_BF16(A, Bm, C) \
  __builtin_amdgcn_wmma_f32_16x16x32_bf16(false, (A), false, (Bm), (short)0, (C), false, false)

// ---------------- grid-wide sense barrier (persistent kernel) ----------------
__device__ __forceinline__ void gridbar(unsigned* cnt, unsigned* gen, unsigned n) {
  __syncthreads();
  if (threadIdx.x == 0) {
    unsigned g = __hip_atomic_load(gen, __ATOMIC_RELAXED, __HIP_MEMORY_SCOPE_AGENT);
    if (__hip_atomic_fetch_add(cnt, 1u, __ATOMIC_ACQ_REL, __HIP_MEMORY_SCOPE_AGENT) == n - 1u) {
      __hip_atomic_store(cnt, 0u, __ATOMIC_RELAXED, __HIP_MEMORY_SCOPE_AGENT);
      __hip_atomic_fetch_add(gen, 1u, __ATOMIC_ACQ_REL, __HIP_MEMORY_SCOPE_AGENT);
    } else {
      while (__hip_atomic_load(gen, __ATOMIC_ACQUIRE, __HIP_MEMORY_SCOPE_AGENT) == g)
        __builtin_amdgcn_s_sleep(2);
    }
  }
  __syncthreads();
}

// ---------------- f32 -> bf16 convert ----------------
__global__ void k_f32_to_bf16(const float* __restrict__ src,
                              unsigned short* __restrict__ dst, int n) {
  int i = blockIdx.x * blockDim.x + threadIdx.x;
  if (i < n) dst[i] = f2bf(src[i]);
}

// ---------------- input-projection GEMM: G = X*Wih^T + bias ----------------
// M = T*B = 16384 rows (row = t*32+b), N = 7168, K = 512.
// One wave computes a 16x64 strip (4 n-tiles sharing the A fragment).
__global__ void __launch_bounds__(256)
k_input_gemm(const unsigned short* __restrict__ xbf,
             const unsigned short* __restrict__ lWih,
             const unsigned short* __restrict__ gWih,
             const float* __restrict__ lbih,
             const float* __restrict__ lbhh,
             const float* __restrict__ gbih,
             float* __restrict__ G) {
  const int lane15 = threadIdx.x & 15;
  const int half   = (threadIdx.x >> 4) & 1;
  const int w      = blockIdx.x * 8 + (threadIdx.x >> 5);   // 0..114687
  const int mt     = w / 112;                               // 1024 m-tiles
  const int nt0    = (w - mt * 112) * 4;                    // first of 4 n-tiles

  // A fragment source: row r = t*32 + b maps to x[b][t][:]
  const int rrow = mt * 16 + lane15;
  const gv4cp Ab =
      (gv4cp)(const v4u*)(xbf + ((size_t)(rrow & 31) * Tt + (rrow >> 5)) * Ii);

  gv4cp w4[4];
  float bias[4];
  int   colg[4];
#pragma unroll
  for (int j = 0; j < 4; ++j) {
    int cg  = (nt0 + j) * 16 + lane15;        // this lane's global output column
    colg[j] = cg;
    if (cg < 4096) {                          // LSTM region: [cell][gate*512+h]
      int cell = cg >> 11;
      int o    = cg & 2047;
      w4[j]    = (gv4cp)(const v4u*)(lWih + ((size_t)cell * 2048 + o) * Ii);
      bias[j]  = lbih[cell * 2048 + o] + lbhh[cell * 2048 + o];
    } else {                                  // GRU region: 4096 + cell*1536 + gate*512 + h
      int cp   = cg - 4096;
      int cell = cp / 1536;
      int o    = cp - cell * 1536;
      w4[j]    = (gv4cp)(const v4u*)(gWih + ((size_t)cell * 1536 + o) * Ii);
      bias[j]  = gbih[cell * 1536 + o];
    }
  }

  v8f acc[4] = {};
  for (int ks = 0; ks < 16; ++ks) {
    const int k0 = ks * 32;
    Frag32 a;                                  // A 16x32: lanes<16: K k0..+15
    const int ai = (k0 >> 3) + half * 2;
    a.u[0] = Ab[ai];
    a.u[1] = Ab[ai + 1];
    const int bi = (k0 >> 3) + half * 2;       // B: 16 contiguous K per lane
#pragma unroll
    for (int j = 0; j < 4; ++j) {
      Frag32 bm;
      bm.u[0] = w4[j][bi];
      bm.u[1] = w4[j][bi + 1];
      acc[j] = WMMA_BF16(a.v, bm.v, acc[j]);
    }
  }

#pragma unroll
  for (int r = 0; r < 8; ++r) {
    const size_t rg = (size_t)(mt * 16 + half * 8 + r) * GCOLS;  // C: M = r (+8 hi-half)
#pragma unroll
    for (int j = 0; j < 4; ++j)
      G[rg + colg[j]] = acc[j][r] + bias[j];
  }
}

// ---------------- persistent recurrent kernel ----------------
// 32 workgroups: wg 0..15 LSTM (cell = (wg>>3)&1, hb = wg&7 -> h cols hb*64..+63)
//                wg 16..31 GRU likewise.
// Whh slice lives in LDS for the whole kernel (<=256KB); the per-step gate slice
// G[t] is DMA'd into LDS by the Tensor Data Mover, overlapped with the GEMM.
__global__ void __launch_bounds__(256)
k_recurrent(const float* __restrict__ G,
            const unsigned short* __restrict__ lWhh,
            const unsigned short* __restrict__ gWhh,
            const float* __restrict__ gbhh,
            unsigned short* __restrict__ hl,   // [2 parity][2 cell][32 b][512]
            unsigned short* __restrict__ hg,
            float* __restrict__ out,           // [32][512][2048]
            unsigned* __restrict__ bar) {
  extern __shared__ unsigned short sW[];
  const int wg    = blockIdx.x;
  const bool isL  = wg < 16;
  const int cell  = (wg >> 3) & 1;
  const int hb    = wg & 7;
  const int tid   = threadIdx.x;
  const int lane15 = tid & 15;
  const int half   = (tid >> 4) & 1;
  const int wave   = tid >> 5;
  const int m      = wave >> 2;   // batch half: rows m*16..m*16+15
  const int sub    = wave & 3;    // 16-col sub-block of the 64-col slice
  const int nG     = isL ? 4 : 3;
  const int gbase  = isL ? cell * 2048 : 4096 + cell * 1536;  // G column base

  // ---- stage Whh slice into LDS: local row lr = gate*64 + j  (row-major, 512 K) ----
  const unsigned short* Wsrc = isL ? (lWhh + (size_t)cell * 2048 * Hh)
                                   : (gWhh + (size_t)cell * 1536 * Hh);
  const int nOwn = isL ? 256 : 192;            // rows (4 or 3 gates x 64)
  for (int e = tid; e < nOwn * 64; e += 256) { // 16B chunks
    int lr = e >> 6, c = e & 63;
    int o  = (lr >> 6) * 512 + hb * 64 + (lr & 63);
    ((v4u*)sW)[lr * 64 + c] =
        ((gv4cp)(const v4u*)(Wsrc + (size_t)o * Hh))[c];
  }
  // ---- zero parity-0 h for owned columns ----
  unsigned short* hbuf = isL ? hl : hg;
  for (int e = tid; e < 32 * 64; e += 256)
    hbuf[(size_t)(cell * 32 + (e >> 6)) * Hh + hb * 64 + (e & 63)] = 0;
  __syncthreads();
  gridbar(bar, bar + 1, NWG_REC);

  const int col = hb * 64 + sub * 16 + lane15;     // owned h column for this lane
  float bhr = 0.f, bhz = 0.f, bhn = 0.f;
  if (!isL) {                                      // GRU recurrent biases (pre-activation hn)
    bhr = gbhh[cell * 1536 + col];
    bhz = gbhh[cell * 1536 + 512 + col];
    bhn = gbhh[cell * 1536 + 1024 + col];
  }
  v8f st = {};                                     // LSTM c-state / GRU h-state tile
  const v4u* sW4 = (const v4u*)sW;
  const int lrb = (sub * 16 + lane15) * 64;        // gate-0 B-row base (16B units)
  const float* gls = (const float*)((const char*)sW + G_LDS_OFF);  // LDS G slice
  const int cc = sub * 16 + lane15;

  for (int t = 0; t < Tt; ++t) {
#if HAVE_TDM
    // ---- TDM: async 3D tile G[t*32..+32][gbase + g*512 + hb*64 ..+64] -> LDS ----
    // tile: dim0 = 64 cols (contig), dim1 = nG gates (stride 512), dim2 = 32 rows
    // (stride 7168); LDS fill order [row][gate][col], f32 elements.
    if (wave == 0) {
      unsigned ldsG = (unsigned)(size_t)(void*)sW + G_LDS_OFF;
      unsigned long long ga = (unsigned long long)(size_t)(const void*)
          (G + ((size_t)t * 32) * GCOLS + gbase + hb * 64);
      v4u g0 = { 1u,                                   // D# count=1, user mode
                 ldsG,                                 // lds_addr
                 (unsigned)ga,                         // global_addr[31:0]
                 (unsigned)(ga >> 32) | (2u << 30) };  // global_addr[56:32] | type=2
      v8i g1 = { (int)0x00020000,                      // data_size = 4B
                 (int)(64u << 16),                     // tensor_dim0 = 64
                 (int)((unsigned)nG << 16),            // tensor_dim1 = nG
                 (int)(64u << 16),                     // tile_dim0 = 64
                 (int)((unsigned)nG | (32u << 16)),    // tile_dim1 = nG, tile_dim2 = 32
                 512,                                  // tensor_dim0_stride = 512
                 (int)((GCOLS & 0xFFFFu) << 16),       // tensor_dim1_stride = 7168 (lo16)
                 0 };                                  // tensor_dim1_stride hi
      v4i g2 = { 32, 0, 0, 0 };                        // tensor_dim2 = 32
      v4i g3 = { 0, 0, 0, 0 };
#if __clang_major__ >= 23
      v8i g4 = { 0, 0, 0, 0, 0, 0, 0, 0 };
      __builtin_amdgcn_tensor_load_to_lds(g0, g1, g2, g3, g4, 0);
#else
      __builtin_amdgcn_tensor_load_to_lds(g0, g1, g2, g3, 0);
#endif
    }
#endif

    const unsigned short* hrd =
        hbuf + (size_t)(t & 1) * 32768 + (size_t)cell * 32 * Hh;
    const gv4cp Ab =
        (gv4cp)(const v4u*)(hrd + (size_t)(m * 16 + lane15) * Hh);

    v8f a0 = {}, a1 = {}, a2 = {}, a3 = {};
    for (int ks = 0; ks < 16; ++ks) {
      const int k0 = ks * 32;
      Frag32 a;
      const int ai = (k0 >> 3) + half * 2;
      a.u[0] = Ab[ai];
      a.u[1] = Ab[ai + 1];
      const int kb   = (k0 >> 3) + half * 2;
      const int bidx = lrb + kb;
      Frag32 b0; b0.u[0] = sW4[bidx];          b0.u[1] = sW4[bidx + 1];
      a0 = WMMA_BF16(a.v, b0.v, a0);
      Frag32 b1; b1.u[0] = sW4[bidx + 4096];   b1.u[1] = sW4[bidx + 4097];
      a1 = WMMA_BF16(a.v, b1.v, a1);
      Frag32 b2; b2.u[0] = sW4[bidx + 8192];   b2.u[1] = sW4[bidx + 8193];
      a2 = WMMA_BF16(a.v, b2.v, a2);
      if (isL) {                                // uniform branch: EXEC stays all-ones
        Frag32 b3; b3.u[0] = sW4[bidx + 12288]; b3.u[1] = sW4[bidx + 12289];
        a3 = WMMA_BF16(a.v, b3.v, a3);
      }
    }

#if HAVE_TDM
    if (wave == 0) __builtin_amdgcn_s_wait_tensorcnt(0);  // DMA done (issuing wave)
    __syncthreads();                                      // publish LDS slice to all waves
#endif

    unsigned short* hwr =
        hbuf + (size_t)((t & 1) ^ 1) * 32768 + (size_t)cell * 32 * Hh;
    if (isL) {
#pragma unroll
      for (int r = 0; r < 8; ++r) {
        const int b = m * 16 + half * 8 + r;
#if HAVE_TDM
        const float* Gr = gls + (size_t)b * 4 * 64;       // [b][gate][64] in LDS
        float gi = Gr[cc]           + a0[r];
        float gf = Gr[64 + cc]      + a1[r];
        float gg = Gr[128 + cc]     + a2[r];
        float go = Gr[192 + cc]     + a3[r];
#else
        const float* Gr = G + ((size_t)t * 32 + b) * GCOLS + cell * 2048;
        float gi = Gr[col]        + a0[r];
        float gf = Gr[512 + col]  + a1[r];
        float gg = Gr[1024 + col] + a2[r];
        float go = Gr[1536 + col] + a3[r];
#endif
        float c = sigm(gf) * st[r] + sigm(gi) * tanhfast(gg);
        st[r] = c;
        float h = sigm(go) * tanhfast(c);
        hwr[(size_t)b * Hh + col] = f2bf(h);
        out[((size_t)b * Tt + t) * 2048 + cell * 1024 + col] = h;
      }
    } else {
#pragma unroll
      for (int r = 0; r < 8; ++r) {
        const int b = m * 16 + half * 8 + r;
#if HAVE_TDM
        const float* Gr = gls + (size_t)b * 3 * 64;       // [b][gate][64] in LDS
        float pir = Gr[cc], piz = Gr[64 + cc], pin = Gr[128 + cc];
#else
        const float* Gr = G + ((size_t)t * 32 + b) * GCOLS + 4096 + cell * 1536;
        float pir = Gr[col], piz = Gr[512 + col], pin = Gr[1024 + col];
#endif
        float rr = sigm(pir + a0[r] + bhr);
        float zz = sigm(piz + a1[r] + bhz);
        float hn = a2[r] + bhn;
        float nn = tanhfast(pin + rr * hn);
        float h  = (1.0f - zz) * nn + zz * st[r];
        st[r] = h;
        hwr[(size_t)b * Hh + col] = f2bf(h);
        out[((size_t)b * Tt + t) * 2048 + cell * 1024 + 512 + col] = h;
      }
    }
#if !HAVE_TDM
    if (t + 1 < Tt)  // pull next step's gate slice toward L2 (global_prefetch_b8)
      __builtin_prefetch(G + ((size_t)(t + 1) * 32 + m * 16 + half * 8) * GCOLS + col, 0, 0);
#endif
    __threadfence();
    gridbar(bar, bar + 1, NWG_REC);
  }
}

// ---------------- host launcher ----------------
extern "C" void kernel_launch(void* const* d_in, const int* in_sizes, int n_in,
                              void* d_out, int out_size, void* d_ws, size_t ws_size,
                              hipStream_t stream) {
  (void)in_sizes; (void)n_in; (void)out_size; (void)ws_size;
  const float* x    = (const float*)d_in[0];
  const float* lWih = (const float*)d_in[1];
  const float* lWhh = (const float*)d_in[2];
  const float* lbih = (const float*)d_in[3];
  const float* lbhh = (const float*)d_in[4];
  const float* gWih = (const float*)d_in[5];
  const float* gWhh = (const float*)d_in[6];
  const float* gbih = (const float*)d_in[7];
  const float* gbhh = (const float*)d_in[8];
  float* out = (float*)d_out;

  char* ws = (char*)d_ws;
  size_t off = 0;
  unsigned*       bar     = (unsigned*)(ws + off);        off += 256;
  unsigned short* xbf     = (unsigned short*)(ws + off);  off += (size_t)Bb * Tt * Ii * 2;
  unsigned short* lWih_bf = (unsigned short*)(ws + off);  off += (size_t)2 * 2048 * 512 * 2;
  unsigned short* lWhh_bf = (unsigned short*)(ws + off);  off += (size_t)2 * 2048 * 512 * 2;
  unsigned short* gWih_bf = (unsigned short*)(ws + off);  off += (size_t)2 * 1536 * 512 * 2;
  unsigned short* gWhh_bf = (unsigned short*)(ws + off);  off += (size_t)2 * 1536 * 512 * 2;
  unsigned short* hl      = (unsigned short*)(ws + off);  off += (size_t)2 * 32768 * 2;
  unsigned short* hg      = (unsigned short*)(ws + off);  off += (size_t)2 * 32768 * 2;
  float*          G       = (float*)(ws + off);           off += (size_t)Tt * Bb * GCOLS * 4;

  (void)hipMemsetAsync(d_ws, 0, 256, stream);   // barrier counters must start at 0

  int n;
  n = Bb * Tt * Ii;   k_f32_to_bf16<<<(n + 255) / 256, 256, 0, stream>>>(x,    xbf,     n);
  n = 2 * 2048 * 512; k_f32_to_bf16<<<(n + 255) / 256, 256, 0, stream>>>(lWih, lWih_bf, n);
  n = 2 * 2048 * 512; k_f32_to_bf16<<<(n + 255) / 256, 256, 0, stream>>>(lWhh, lWhh_bf, n);
  n = 2 * 1536 * 512; k_f32_to_bf16<<<(n + 255) / 256, 256, 0, stream>>>(gWih, gWih_bf, n);
  n = 2 * 1536 * 512; k_f32_to_bf16<<<(n + 255) / 256, 256, 0, stream>>>(gWhh, gWhh_bf, n);

  // 1024 m-tiles * 112 n-groups = 114688 waves / 8 per block
  k_input_gemm<<<14336, 256, 0, stream>>>(xbf, lWih_bf, gWih_bf, lbih, lbhh, gbih, G);

  // persistent recurrent loop: 32 co-resident workgroups,
  // 256KB weights + 32KB TDM gate-slice buffer = 288KB dynamic LDS
  k_recurrent<<<NWG_REC, 256, G_LDS_OFF + 32768, stream>>>(G, lWhh_bf, gWhh_bf, gbhh,
                                                           hl, hg, out, bar);
}
